// FoItttLinear_26474178412911
// MI455X (gfx1250) — matrix operand order repaired and checked by reading, not compiled, not run on hardware
//
#include <hip/hip_runtime.h>

#define BSZ   4
#define SEQ   2048
#define DIN   4096
#define DOUT  4096
#define RNK   128
#define KT    64          // K-tile per barrier (two 16x16x32 k-steps)

typedef __attribute__((ext_vector_type(16))) __bf16 v16bf;
typedef __attribute__((ext_vector_type(8)))  float  v8f;

// Pack two floats into two bf16 (round-half-up) with one v_perm_b32:
// result[15:0] = hi16(a + 0x8000), result[31:16] = hi16(b + 0x8000).
// Selector bytes 2,3 pick from src1 (ua), bytes 6,7 pick from src0 (ub).
__device__ __forceinline__ unsigned pack2bf(float a, float b) {
    unsigned ua = __builtin_bit_cast(unsigned, a) + 0x8000u;
    unsigned ub = __builtin_bit_cast(unsigned, b) + 0x8000u;
    return __builtin_amdgcn_perm(ub, ua, 0x07060302u);
}

// ---------------------------------------------------------------------------
// Kernel 1: A[b,r,i] = 0.01*exp(64*log_lr[r,i]) * state[b,r,i] + w_bsp[r,i]
// ---------------------------------------------------------------------------
__global__ __launch_bounds__(256) void prep_A(const float* __restrict__ log_lr,
                                              const float* __restrict__ state,
                                              const float* __restrict__ w_bsp,
                                              float* __restrict__ A) {
    int idx = blockIdx.x * 256 + threadIdx.x;          // over BSZ*RNK*DIN
    int j   = idx & (RNK * DIN - 1);                   // r*DIN + i
    float l = 0.01f * __expf(log_lr[j] * 64.0f);       // sqrt(4096) == 64
    A[idx]  = __builtin_fmaf(l, state[idx], w_bsp[j]);
}

// ---------------------------------------------------------------------------
// Global fetch of one 128x64 X tile + one 128x64 W tile into registers.
// 256 threads: thread t -> row t/2, 32 consecutive cols at (t&1)*32.
// ---------------------------------------------------------------------------
struct Fetch {
    float4 x[8];
    float4 w[8];
};

__device__ __forceinline__ Fetch fetch_tiles(const float* __restrict__ X, int ldx, int m_base,
                                             const float* __restrict__ W, int ldw, int n_base,
                                             int kb) {
    Fetch f;
    const int t   = threadIdx.x;
    const int row = t >> 1;
    const int col = (t & 1) << 5;
    const float4* px = (const float4*)(X + (size_t)(m_base + row) * ldx + kb + col);
    const float4* pw = (const float4*)(W + (size_t)(n_base + row) * ldw + kb + col);
#pragma unroll
    for (int q = 0; q < 8; ++q) f.x[q] = px[q];
#pragma unroll
    for (int q = 0; q < 8; ++q) f.w[q] = pw[q];
    return f;
}

// cvt fp32->bf16 (packed via v_perm) and store one staged tile half into LDS
__device__ __forceinline__ void stage_one(unsigned short* __restrict__ sm,
                                          const float4* __restrict__ v) {
    const int t   = threadIdx.x;
    const int row = t >> 1;
    const int col = (t & 1) << 5;
    unsigned h[16];
#pragma unroll
    for (int q = 0; q < 8; ++q) {
        h[q * 2 + 0] = pack2bf(v[q].x, v[q].y);
        h[q * 2 + 1] = pack2bf(v[q].z, v[q].w);
    }
    uint4* d = (uint4*)(sm + row * KT + col);
#pragma unroll
    for (int q = 0; q < 4; ++q) d[q] = ((const uint4*)h)[q];
}

// ---------------------------------------------------------------------------
// Per-wave compute of one 64-K LDS tile: 2 A-frags x 4 B-frags, 16 WMMAs.
// All fragment loads are issued before the WMMA chain of each k-step so the
// compiler can merge the dscnt waits.
// ---------------------------------------------------------------------------
__device__ __forceinline__ void compute_tile(const unsigned short* __restrict__ sx,
                                             const unsigned short* __restrict__ sw,
                                             int wm, int wn, int kh, int ln,
                                             v8f acc[2][4]) {
#pragma unroll
    for (int s = 0; s < 2; ++s) {
        const int ko = s * 32;
        v16bf a[2], bf[4];
#pragma unroll
        for (int i = 0; i < 2; ++i) {
            // A layout (16-bit, 16x32): lane row M=ln, K = kh*8+0..7 and 16+kh*8+0..7
            const unsigned short* ap = sx + (wm * 32 + i * 16 + ln) * KT + ko + kh * 8;
            ((uint4*)&a[i])[0] = *(const uint4*)(ap);
            ((uint4*)&a[i])[1] = *(const uint4*)(ap + 16);
        }
#pragma unroll
        for (int j = 0; j < 4; ++j) {
            // B layout (32x16): lane col N=ln, K = kh*8+0..7 and 16+kh*8+0..7
            const unsigned short* bp = sw + (wn * 64 + j * 16 + ln) * KT + ko + kh * 8;
            ((uint4*)&bf[j])[0] = *(const uint4*)(bp);
            ((uint4*)&bf[j])[1] = *(const uint4*)(bp + 16);
        }
#pragma unroll
        for (int i = 0; i < 2; ++i)
#pragma unroll
            for (int j = 0; j < 4; ++j)
                acc[i][j] = __builtin_amdgcn_wmma_f32_16x16x32_bf16(
                    false, a[i], false, bf[j], (short)0, acc[i][j], false, false);
    }
}

// ---------------------------------------------------------------------------
// Two-segment pipelined WMMA GEMM:
//   Y[b] = X1[b] @ W1[b]^T (+ X2[b] @ W2[b]^T) (+ bias)
// Block: 256 threads (8 waves, 4Mx2N wave grid). Tile 128(M) x 128(N).
// Double-buffered LDS (64 KB), global loads in flight during WMMA.
// ---------------------------------------------------------------------------
__global__ __launch_bounds__(256) void gemm2seg(
    const float* __restrict__ X1, size_t sX1, int ld1, int K1,
    const float* __restrict__ W1, size_t sW1, int ldw1,
    const float* __restrict__ X2, size_t sX2, int ld2, int K2,
    const float* __restrict__ W2, size_t sW2, int ldw2,
    const float* __restrict__ bias,
    float* __restrict__ Y, size_t sY, int ldy)
{
    __shared__ __align__(16) unsigned short smX[2][128 * KT];   // 2 x 16 KB
    __shared__ __align__(16) unsigned short smW[2][128 * KT];   // 2 x 16 KB

    const int b      = blockIdx.z;
    const int m_base = blockIdx.y * 128;
    const int n_base = blockIdx.x * 128;
    const int lane   = threadIdx.x & 31;
    const int wave   = threadIdx.x >> 5;
    const int wm     = wave >> 1;      // 0..3 : 32-row band
    const int wn     = wave & 1;       // 0..1 : 64-col band
    const int kh     = lane >> 4;
    const int ln     = lane & 15;

    v8f acc[2][4];
#pragma unroll
    for (int i = 0; i < 2; ++i)
#pragma unroll
        for (int j = 0; j < 4; ++j)
            acc[i][j] = (v8f){0.f, 0.f, 0.f, 0.f, 0.f, 0.f, 0.f, 0.f};

#pragma unroll 1
    for (int seg = 0; seg < 2; ++seg) {
        const float* Xs  = seg ? X2 : X1;
        const float* Ws  = seg ? W2 : W1;
        const int    ldx = seg ? ld2 : ld1;
        const int    ldw = seg ? ldw2 : ldw1;
        const int    K   = seg ? K2 : K1;
        if (K <= 0 || Xs == nullptr) continue;            // uniform across block
        const float* Xb = Xs + (size_t)b * (seg ? sX2 : sX1);
        const float* Wb = Ws + (size_t)b * (seg ? sW2 : sW1);
        const int nt = K / KT;

        __syncthreads();                                  // buffers free for reuse
        {
            Fetch f0 = fetch_tiles(Xb, ldx, m_base, Wb, ldw, n_base, 0);
            stage_one(smX[0], f0.x);
            stage_one(smW[0], f0.w);
        }

#pragma unroll 1
        for (int kt = 0; kt < nt; ++kt) {
            const int  cur  = kt & 1;
            const bool more = (kt + 1) < nt;
            Fetch fn;
            if (more)                                     // issue next global loads
                fn = fetch_tiles(Xb, ldx, m_base, Wb, ldw, n_base, (kt + 1) * KT);
            __syncthreads();                              // cur buffer staged
            compute_tile(smX[cur], smW[cur], wm, wn, kh, ln, acc);  // loads in flight
            if (more) {
                stage_one(smX[1 - cur], fn.x);
                stage_one(smW[1 - cur], fn.w);
            }
        }
    }

    // Epilogue: D layout — lanes 0-15: M = v, lanes 16-31: M = 8+v; N = ln.
    float* Yb = Y + (size_t)b * sY;
#pragma unroll
    for (int i = 0; i < 2; ++i) {
        const int row0 = m_base + wm * 32 + i * 16 + kh * 8;
#pragma unroll
        for (int j = 0; j < 4; ++j) {
            const int col = n_base + wn * 64 + j * 16 + ln;
            const float bv = bias ? bias[col] : 0.0f;
#pragma unroll
            for (int v = 0; v < 8; ++v)
                Yb[(size_t)(row0 + v) * ldy + col] = acc[i][j][v] + bv;
        }
    }
}

// ---------------------------------------------------------------------------
// Launch
// inputs: 0 x, 1 log_lr, 2 state, 3 grad_buffer(unused), 4 w_bsp, 5 w_out,
//         6 w_lin, 7 b_lin
// ---------------------------------------------------------------------------
extern "C" void kernel_launch(void* const* d_in, const int* in_sizes, int n_in,
                              void* d_out, int out_size, void* d_ws, size_t ws_size,
                              hipStream_t stream) {
    const float* x      = (const float*)d_in[0];
    const float* log_lr = (const float*)d_in[1];
    const float* state  = (const float*)d_in[2];
    const float* w_bsp  = (const float*)d_in[4];
    const float* w_out  = (const float*)d_in[5];
    const float* w_lin  = (const float*)d_in[6];
    const float* b_lin  = (const float*)d_in[7];
    float* y = (float*)d_out;

    float* A = (float*)d_ws;                          // [BSZ, RNK, DIN] fp32, 8 MB
    float* z = A + (size_t)BSZ * RNK * DIN;           // [BSZ, SEQ, RNK] fp32, 4 MB

    // 1) effective state weights
    prep_A<<<(BSZ * RNK * DIN) / 256, 256, 0, stream>>>(log_lr, state, w_bsp, A);

    // 2) z[b] = X[b] @ A[b]^T    (M=2048, N=128, K=4096)
    gemm2seg<<<dim3(1, SEQ / 128, BSZ), 256, 0, stream>>>(
        x, (size_t)SEQ * DIN, DIN, DIN,
        A, (size_t)RNK * DIN, DIN,
        nullptr, 0, 0, 0,
        nullptr, 0, 0,
        nullptr,
        z, (size_t)SEQ * RNK, RNK);

    // 3) y[b] = X[b] @ w_lin^T + z[b] @ w_out^T + b_lin  (M=2048, N=4096)
    gemm2seg<<<dim3(DOUT / 128, SEQ / 128, BSZ), 256, 0, stream>>>(
        x, (size_t)SEQ * DIN, DIN, DIN,
        w_lin, 0, DIN,
        z, (size_t)SEQ * RNK, RNK, RNK,
        w_out, 0, RNK,
        b_lin,
        y, (size_t)SEQ * DOUT, DOUT);
}
